// Decoder_88519275971172
// MI455X (gfx1250) — matrix operand order repaired and checked by reading, not compiled, not run on hardware
//
#include <hip/hip_runtime.h>
#include <hip/hip_bf16.h>
#include <math.h>

// ---------------------------------------------------------------------------
// CDNA5 (gfx1250) wave32 WMMA bf16 kernels for an attention-decoder step.
// Wave tile: 32(M) x 64(N), K-step 32 -> 8 v_wmma_f32_16x16x32_bf16 per step.
// A-side operands pre-converted to bf16 (pure b128 loads in hot loop);
// B-side (weights) converted fp32->bf16 in-register via packed cvt.
// ---------------------------------------------------------------------------

typedef __attribute__((ext_vector_type(16))) __bf16 v16bf;
typedef __attribute__((ext_vector_type(16))) float  v16f;
typedef __attribute__((ext_vector_type(8)))  float  v8f;

// fp32 -> bf16 bits (rne) for scalar epilogue/elementwise paths
__device__ __forceinline__ unsigned short f2bfu(float f) {
  union { float f; unsigned u; } x; x.f = f;
  unsigned r = x.u + 0x7FFFu + ((x.u >> 16) & 1u);
  return (unsigned short)(r >> 16);
}

__device__ __forceinline__ float sigmoidf_(float x) { return 1.0f / (1.0f + expf(-x)); }

// A-matrix 16x32 bf16 tile, A already bf16 in memory.
// ISA layout: lane L<16 -> M=L, K={0..7,16..23}; lane L>=16 -> M=L-16,
// K={8..15,24..31}.  ka0 = 8*half, ka1 = ka0+16.  Two 16B loads, no VALU.
__device__ __forceinline__ v16bf load_a_bf(const unsigned short* __restrict__ arow,
                                           int k, int ka0, int ka1) {
  union { uint4 q[2]; v16bf v; } u;
  u.q[0] = *(const uint4*)(arow + k + ka0);
  u.q[1] = *(const uint4*)(arow + k + ka1);
  return u.v;
}

// A tile from fp32 source (energy kernel): 4x b128 load + packed cvt.
__device__ __forceinline__ v16bf load_a_cvt(const float* __restrict__ arow,
                                            int k, int ka0, int ka1) {
  union { float4 q[4]; v16f v; } t;
  t.q[0] = *(const float4*)(arow + k + ka0);
  t.q[1] = *(const float4*)(arow + k + ka0 + 4);
  t.q[2] = *(const float4*)(arow + k + ka1);
  t.q[3] = *(const float4*)(arow + k + ka1 + 4);
  return __builtin_convertvector(t.v, v16bf);
}

// B-matrix 32x16 bf16 tile from fp32 weights: lane L<16 -> N=L, K={0..7,16..23};
// lane L>=16 -> N=L-16, K={8..15,24..31}.  brow points at W[n]*ldb + k + 8*half.
__device__ __forceinline__ v16bf load_b_cvt(const float* __restrict__ brow) {
  union { float4 q[4]; v16f v; } t;
#pragma unroll
  for (int i = 0; i < 4; ++i) t.q[i] = *(const float4*)(brow + 4 * i + (i >> 1) * 8);
  return __builtin_convertvector(t.v, v16bf);
}

// ---------------------------------------------------------------------------
// C[M,N] = (accum ? C : 0) + A_bf16[M,K] @ W_f32[N,K]^T + bias[n]
// One wave owns a 32x64 tile (8 accumulators; each B tile feeds 2 WMMAs).
// Requires M%32==0, N%64==0, K%32==0.
// ---------------------------------------------------------------------------
__global__ __launch_bounds__(256)
void wmma_gemm(const unsigned short* __restrict__ A, int lda,
               const float* __restrict__ W, int ldb,
               const float* __restrict__ bias,
               float* __restrict__ C, int ldc,
               int M, int N, int K, int accum) {
  const int wid  = (blockIdx.x * blockDim.x + threadIdx.x) >> 5;
  const int lane = threadIdx.x & 31;
  const int nt64 = N >> 6;
  const int mt   = wid / nt64;
  const int nt   = wid % nt64;
  if (mt * 32 >= M) return;

  const int m0   = mt << 5;
  const int n0   = nt << 6;
  const int half = lane >> 4;
  const int lm   = lane & 15;
  const int ka0  = half * 8;
  const int ka1  = ka0 + 16;
  const int kb   = half * 8;

  const unsigned short* __restrict__ arow0 = A + (size_t)(m0 + lm) * lda;
  const unsigned short* __restrict__ arow1 = A + (size_t)(m0 + 16 + lm) * lda;

  v8f acc[8] = {};
  for (int k = 0; k < K; k += 32) {
    v16bf a0 = load_a_bf(arow0, k, ka0, ka1);
    v16bf a1 = load_a_bf(arow1, k, ka0, ka1);
#pragma unroll
    for (int t = 0; t < 4; ++t) {
      const float* brow = W + (size_t)(n0 + t * 16 + lm) * ldb + k + kb;
      v16bf b = load_b_cvt(brow);
      acc[t]     = __builtin_amdgcn_wmma_f32_16x16x32_bf16(
          false, a0, false, b, (short)0, acc[t], false, false);
      acc[4 + t] = __builtin_amdgcn_wmma_f32_16x16x32_bf16(
          false, a1, false, b, (short)0, acc[4 + t], false, false);
    }
  }

#pragma unroll
  for (int t = 0; t < 4; ++t) {
    const int   n  = n0 + t * 16 + lm;
    const float bv = bias ? bias[n] : 0.0f;
#pragma unroll
    for (int r = 0; r < 8; ++r) {
      const int m0r = m0 + half * 8 + r;        // rows of acc[t]
      const int m1r = m0r + 16;                 // rows of acc[4+t]
      float v0 = acc[t][r] + bv;
      float v1 = acc[4 + t][r] + bv;
      if (accum) {
        v0 += C[(size_t)m0r * ldc + n];
        v1 += C[(size_t)m1r * ldc + n];
      }
      C[(size_t)m0r * ldc + n] = v0;
      C[(size_t)m1r * ldc + n] = v1;
    }
  }
}

// ---------------------------------------------------------------------------
// Fused attention energy + score contraction (energy never materialized):
//   scores[m] = Vb + sum_n tanh( enc[m,:]@W1[n,:] + W1b[n] + hproj[m/S][n] ) * Vw[n]
// One wave owns 32 rows and sweeps N=1024 in 16 tiles of 64; the Vw-weighted
// row sum stays in registers -> deterministic, atomic-free.  Saves ~270 MB of
// HBM round-trip for the [32768,1024] energy tensor.
// ---------------------------------------------------------------------------
__global__ __launch_bounds__(256)
void wmma_energy(const float* __restrict__ enc,     // [32768,1024]
                 const float* __restrict__ W1,      // [1024,1024]
                 const float* __restrict__ W1b,
                 const float* __restrict__ hproj,   // [128,1024]
                 const float* __restrict__ Vw,      // [1024]
                 const float* __restrict__ Vb,
                 float* __restrict__ scores) {      // [32768]
  const int wid  = (blockIdx.x * blockDim.x + threadIdx.x) >> 5;
  const int lane = threadIdx.x & 31;
  if (wid >= 1024) return;               // 32768 rows / 32
  const int m0   = wid << 5;
  const int half = lane >> 4;
  const int lm   = lane & 15;
  const int ka0  = half * 8;
  const int ka1  = ka0 + 16;
  const int kb   = half * 8;
  const int bb   = m0 >> 8;              // batch (S=256; 32-row group in-batch)

  const float* __restrict__ arow0 = enc + (size_t)(m0 + lm) * 1024;
  const float* __restrict__ arow1 = enc + (size_t)(m0 + 16 + lm) * 1024;

  float rowsum[16];
#pragma unroll
  for (int r = 0; r < 16; ++r) rowsum[r] = 0.0f;

  for (int nt = 0; nt < 16; ++nt) {
    const int n0 = nt << 6;
    v8f acc[8] = {};
    for (int k = 0; k < 1024; k += 32) {
      v16bf a0 = load_a_cvt(arow0, k, ka0, ka1);
      v16bf a1 = load_a_cvt(arow1, k, ka0, ka1);
#pragma unroll
      for (int t = 0; t < 4; ++t) {
        const float* brow = W1 + (size_t)(n0 + t * 16 + lm) * 1024 + k + kb;
        v16bf b = load_b_cvt(brow);
        acc[t]     = __builtin_amdgcn_wmma_f32_16x16x32_bf16(
            false, a0, false, b, (short)0, acc[t], false, false);
        acc[4 + t] = __builtin_amdgcn_wmma_f32_16x16x32_bf16(
            false, a1, false, b, (short)0, acc[4 + t], false, false);
      }
    }
#pragma unroll
    for (int t = 0; t < 4; ++t) {
      const int   n  = n0 + t * 16 + lm;
      const float ad = W1b[n] + hproj[(size_t)bb * 1024 + n];
      const float vn = Vw[n];
#pragma unroll
      for (int r = 0; r < 8; ++r) {
        rowsum[r]     += tanhf(acc[t][r]     + ad) * vn;
        rowsum[8 + r] += tanhf(acc[4 + t][r] + ad) * vn;
      }
    }
  }

  // reduce the 16 column-partials across each 16-lane half (xor stays in-half)
#pragma unroll
  for (int r = 0; r < 16; ++r) {
    float s = rowsum[r];
#pragma unroll
    for (int off = 1; off < 16; off <<= 1) s += __shfl_xor(s, off, 32);
    if (lm == 0) scores[m0 + (r >> 3) * 16 + half * 8 + (r & 7)] = s + Vb[0];
  }
}

// ---------------------------------------------------------------------------
// Softmax over S=256 + context = attn @ enc; context written as bf16 straight
// into x0[:,512:] and xcat[:,1024:].  One block per batch row.
// ---------------------------------------------------------------------------
__global__ __launch_bounds__(256)
void softmax_context(const float* __restrict__ scores,
                     const float* __restrict__ enc,
                     float* __restrict__ attn_out,          // [128,256]
                     unsigned short* __restrict__ x0,       // [128,1536] bf16
                     unsigned short* __restrict__ xcat) {   // [128,2048] bf16
  const int b = blockIdx.x, tid = threadIdx.x;
  __shared__ float red[256];
  __shared__ float att[256];

  float v = scores[b * 256 + tid];
  red[tid] = v; __syncthreads();
  for (int s = 128; s > 0; s >>= 1) {
    if (tid < s) red[tid] = fmaxf(red[tid], red[tid + s]);
    __syncthreads();
  }
  const float mx = red[0]; __syncthreads();
  float e = expf(v - mx);
  red[tid] = e; __syncthreads();
  for (int s = 128; s > 0; s >>= 1) {
    if (tid < s) red[tid] += red[tid + s];
    __syncthreads();
  }
  const float a = e / red[0];
  att[tid] = a;
  attn_out[b * 256 + tid] = a;
  __syncthreads();

  const int h0 = tid * 4;
  float c0 = 0.f, c1 = 0.f, c2 = 0.f, c3 = 0.f;
  for (int s = 0; s < 256; ++s) {
    const float w = att[s];
    const float4 ev = *(const float4*)&enc[((size_t)b * 256 + s) * 1024 + h0];
    c0 += w * ev.x; c1 += w * ev.y; c2 += w * ev.z; c3 += w * ev.w;
  }
  unsigned short b0 = f2bfu(c0), b1 = f2bfu(c1), b2 = f2bfu(c2), b3 = f2bfu(c3);
  unsigned short* px = x0   + (size_t)b * 1536 + 512 + h0;
  unsigned short* pc = xcat + (size_t)b * 2048 + 1024 + h0;
  px[0] = b0; px[1] = b1; px[2] = b2; px[3] = b3;
  pc[0] = b0; pc[1] = b1; pc[2] = b2; pc[3] = b3;
}

__global__ void embed_kernel(const int* __restrict__ tok,
                             const float* __restrict__ emb,
                             unsigned short* __restrict__ x0) {  // bf16
  const int idx = blockIdx.x * blockDim.x + threadIdx.x;
  if (idx >= 128 * 512) return;
  const int b = idx >> 9, e = idx & 511;
  x0[(size_t)b * 1536 + e] = f2bfu(emb[(size_t)tok[b] * 512 + e]);
}

__global__ void cvt_f32_bf16(const float* __restrict__ src,
                             unsigned short* __restrict__ dst, int n) {
  const int idx = blockIdx.x * blockDim.x + threadIdx.x;
  if (idx < n) dst[idx] = f2bfu(src[idx]);
}

__global__ void lstm_elem(const float* __restrict__ gates,   // [128,4096] i|f|g|o
                          const float* __restrict__ c_in,    // [128,1024]
                          float* __restrict__ h_out,
                          float* __restrict__ c_out,
                          unsigned short* __restrict__ h_bf, // bf16 copy of h
                          int h_bf_ld) {
  const int idx = blockIdx.x * blockDim.x + threadIdx.x;
  if (idx >= 128 * 1024) return;
  const int b = idx >> 10, j = idx & 1023;
  const float* g = gates + (size_t)b * 4096;
  const float i  = sigmoidf_(g[j]);
  const float f  = sigmoidf_(g[1024 + j]);
  const float gg = tanhf(g[2048 + j]);
  const float o  = sigmoidf_(g[3072 + j]);
  const float cn = f * c_in[idx] + i * gg;
  const float hn = o * tanhf(cn);
  h_out[idx] = hn;
  c_out[idx] = cn;
  h_bf[(size_t)b * h_bf_ld + j] = f2bfu(hn);
}

__global__ __launch_bounds__(256)
void logsoftmax_inplace(float* __restrict__ logits, int N) {
  const int b = blockIdx.x, tid = threadIdx.x;
  float* row = logits + (size_t)b * N;
  __shared__ float red[256];
  float m = -3.0e38f;
  for (int i = tid; i < N; i += 256) m = fmaxf(m, row[i]);
  red[tid] = m; __syncthreads();
  for (int s = 128; s > 0; s >>= 1) {
    if (tid < s) red[tid] = fmaxf(red[tid], red[tid + s]);
    __syncthreads();
  }
  m = red[0]; __syncthreads();
  float sum = 0.f;
  for (int i = tid; i < N; i += 256) sum += expf(row[i] - m);
  red[tid] = sum; __syncthreads();
  for (int s = 128; s > 0; s >>= 1) {
    if (tid < s) red[tid] += red[tid + s];
    __syncthreads();
  }
  const float lse = m + logf(red[0]);
  for (int i = tid; i < N; i += 256) row[i] -= lse;
}

// ---------------------------------------------------------------------------

static inline void launch_gemm(const unsigned short* A, int lda, const float* W,
                               int ldb, const float* bias, float* C, int ldc,
                               int M, int N, int K, int accum, hipStream_t s) {
  const int waves  = (M / 32) * (N / 64);
  const int blocks = (waves + 7) / 8;
  wmma_gemm<<<blocks, 256, 0, s>>>(A, lda, W, ldb, bias, C, ldc, M, N, K, accum);
}

extern "C" void kernel_launch(void* const* d_in, const int* in_sizes, int n_in,
                              void* d_out, int out_size, void* d_ws, size_t ws_size,
                              hipStream_t stream) {
  const int*   tok  = (const int*)  d_in[0];
  const float* hid  = (const float*)d_in[1];   // [2,128,1024]
  const float* cel  = (const float*)d_in[2];   // [2,128,1024]
  const float* enc  = (const float*)d_in[3];   // [128,256,1024]
  const float* emb  = (const float*)d_in[4];   // [32000,512]
  const float* W1w  = (const float*)d_in[5];
  const float* W1b  = (const float*)d_in[6];
  const float* W2w  = (const float*)d_in[7];
  const float* W2b  = (const float*)d_in[8];
  const float* Vw   = (const float*)d_in[9];
  const float* Vb   = (const float*)d_in[10];
  const float* wih0 = (const float*)d_in[11];  // [4096,1536]
  const float* whh0 = (const float*)d_in[12];  // [4096,1024]
  const float* bih0 = (const float*)d_in[13];
  const float* bhh0 = (const float*)d_in[14];
  const float* wih1 = (const float*)d_in[15];  // [4096,1024]
  const float* whh1 = (const float*)d_in[16];
  const float* bih1 = (const float*)d_in[17];
  const float* bhh1 = (const float*)d_in[18];
  const float* fcw  = (const float*)d_in[19];  // [32000,2048]
  const float* fcb  = (const float*)d_in[20];

  const int B = 128, H = 1024, V = 32000, BH = B * H;

  // outputs: prediction [128,32000] | hidden [2,128,1024] | cell | attn [128,256,1]
  float* pred    = (float*)d_out;
  float* hidOut  = pred + (size_t)B * V;
  float* celOut  = hidOut + 2 * BH;
  float* attnOut = celOut + 2 * BH;

  // workspace: fp32 region, then bf16 region (all sub-buffers 16B aligned)
  float* wsf    = (float*)d_ws;
  float* hproj  = wsf;                    // [128,1024]
  float* scores = hproj + 128 * 1024;     // [32768]
  float* gates  = scores + 32768;         // [128,4096]
  unsigned short* wsb    = (unsigned short*)(gates + 128 * 4096);
  unsigned short* x0_bf  = wsb;                    // [128,1536]  emb | ctx
  unsigned short* hid_bf = x0_bf  + 128 * 1536;    // [2,128,1024]
  unsigned short* h0_bf  = hid_bf + 2 * BH;        // [128,1024]
  unsigned short* xcat_bf= h0_bf  + BH;            // [128,2048]  h1 | ctx

  // 1. bf16 copies of A-side inputs
  cvt_f32_bf16<<<(2 * BH + 255) / 256, 256, 0, stream>>>(hid, hid_bf, 2 * BH);
  embed_kernel<<<(B * 512 + 255) / 256, 256, 0, stream>>>(tok, emb, x0_bf);

  // 2. hproj = h_top @ W2^T + W2b   (h_top = hidden[1])
  launch_gemm(hid_bf + BH, H, W2w, H, W2b, hproj, H, B, H, H, 0, stream);

  // 3. fused energy/tanh/V contraction -> scores  (1024 waves)
  wmma_energy<<<128, 256, 0, stream>>>(enc, W1w, W1b, hproj, Vw, Vb, scores);

  // 4. softmax + context (fills attn, x0_bf[:,512:], xcat_bf[:,1024:])
  softmax_context<<<B, 256, 0, stream>>>(scores, enc, attnOut, x0_bf, xcat_bf);

  // 5. LSTM cell 0
  launch_gemm(x0_bf, 1536, wih0, 1536, bih0, gates, 4096, B, 4096, 1536, 0, stream);
  launch_gemm(hid_bf, H, whh0, H, bhh0, gates, 4096, B, 4096, H, 1, stream);
  lstm_elem<<<(BH + 255) / 256, 256, 0, stream>>>(gates, cel, hidOut, celOut,
                                                  h0_bf, 1024);

  // 6. LSTM cell 1 (h1 -> d_out and xcat_bf[:, :1024])
  launch_gemm(h0_bf, H, wih1, H, bih1, gates, 4096, B, 4096, H, 0, stream);
  launch_gemm(hid_bf + BH, H, whh1, H, bhh1, gates, 4096, B, 4096, H, 1, stream);
  lstm_elem<<<(BH + 255) / 256, 256, 0, stream>>>(gates, cel + BH, hidOut + BH,
                                                  celOut + BH, xcat_bf, 2048);

  // 7. logits = xcat @ fc_w^T + fc_b -> pred (in place), then log_softmax
  launch_gemm(xcat_bf, 2048, fcw, 2048, fcb, pred, V, B, V, 2048, 0, stream);
  logsoftmax_inplace<<<B, 256, 0, stream>>>(pred, V);
}
